// MultiHeadAttention_59407987638753
// MI455X (gfx1250) — matrix operand order repaired and checked by reading, not compiled, not run on hardware
//
#include <hip/hip_runtime.h>
#include <hip/hip_bf16.h>
#include <stdint.h>

typedef __attribute__((ext_vector_type(16))) _Float16 v16h;
typedef __attribute__((ext_vector_type(8)))  _Float16 v8h;
typedef __attribute__((ext_vector_type(8)))  float    v8f;

#define B_  4
#define T_  2048
#define C_  1024
#define H_  16
#define HD_ 64

static __device__ __forceinline__ v8f wmma32(v16h a, v16h b, v8f c) {
    // D = A(16x32 f16) * B(32x16 f16) + C(16x16 f32)
    return __builtin_amdgcn_wmma_f32_16x16x32_f16(false, a, false, b, (short)0, c,
                                                  false, false);
}

static __device__ __forceinline__ v16h combine16(v8h lo, v8h hi) {
    v16h r;
#pragma unroll
    for (int i = 0; i < 8; i++) { r[i] = lo[i]; r[i + 8] = hi[i]; }
    return r;
}

// gfx1250 async global->LDS 16B copy (ASYNCcnt-tracked, ISA §15.18.3 op 98)
static __device__ __forceinline__ void async_cp16(uint32_t lds_off, const void* g) {
    uint64_t ga = (uint64_t)(uintptr_t)g;
    asm volatile("global_load_async_to_lds_b128 %0, %1, off"
                 :: "v"(lds_off), "v"(ga) : "memory");
}
// async loads complete in order: <=4 outstanding => oldest tile's 4 landed
static __device__ __forceinline__ void async_wait4() {
    asm volatile("s_wait_asynccnt 0x4" ::: "memory");
}
static __device__ __forceinline__ void async_wait0() {
    asm volatile("s_wait_asynccnt 0x0" ::: "memory");
}

// ---------------------------------------------------------------------------
// Kernel 1: per-head QKV projections.  q,k -> f16 [BH,T,64] (q pre-scaled by
// 1/sqrt(hd)); v -> f16 transposed [BH,64,T].  The head's 64x64 weight matrix
// is staged once per block in LDS, transposed, so each B-fragment is one
// contiguous 32B ds read.
// ---------------------------------------------------------------------------
__global__ void k_qkv(const float* __restrict__ x,
                      const float* __restrict__ Wq,
                      const float* __restrict__ Wk,
                      const float* __restrict__ Wv,
                      _Float16* __restrict__ qb,
                      _Float16* __restrict__ kb,
                      _Float16* __restrict__ vtb)
{
    __shared__ __align__(16) _Float16 wldsT[HD_ * HD_];   // 8 KB, W^T (e,d)
    const int tid  = threadIdx.x;
    const int lane = tid & 31;
    const int wave = tid >> 5;
    const int task = blockIdx.x * 8 + wave;    // 3 * 64 * 128 tasks
    const int mt = task % 128;                 // (p,bh) uniform per block
    const int bh = (task / 128) % 64;
    const int p  = task / (128 * 64);          // 0=q 1=k 2=v
    const int b  = bh >> 4, h = bh & 15;
    const int t0 = mt * 16;
    const int m    = lane & 15;
    const int n    = lane & 15;
    const int half = lane >> 4;

    const float* W = (p == 0 ? Wq : (p == 1 ? Wk : Wv)) + (size_t)h * HD_ * HD_;

    // cooperative transpose-stage W into LDS as f16 (16 elems/thread)
#pragma unroll
    for (int i = 0; i < 16; i++) {
        const int l = tid * 16 + i;            // l = d*64 + e
        const int d = l >> 6, e = l & 63;
        wldsT[e * HD_ + d] = (_Float16)W[l];
    }
    __syncthreads();

    // A fragments from fp32 x (row m of the tile, head slice)
    const float* xr = x + ((size_t)(b * T_ + t0 + m) * C_ + h * HD_);
    v16h a[2];
#pragma unroll
    for (int kd = 0; kd < 2; kd++) {
        const int d0 = kd * 32;
        v16h af;
#pragma unroll
        for (int i = 0; i < 8; i++) {
            af[i]     = (_Float16)xr[d0 + half * 16 + i];
            af[i + 8] = (_Float16)xr[d0 + half * 16 + 8 + i];
        }
        a[kd] = af;
    }

#pragma unroll
    for (int nt = 0; nt < 4; nt++) {
        const int e0 = nt * 16;
        v8f c = {};
#pragma unroll
        for (int kd = 0; kd < 2; kd++) {
            v16h bf = *(const v16h*)(wldsT + (e0 + n) * HD_ + kd * 32 + half * 16);
            c = wmma32(a[kd], bf, c);
        }
        if (p == 0) {
#pragma unroll
            for (int j = 0; j < 8; j++) c[j] *= 0.125f;  // 1/sqrt(64)
        }
        if (p < 2) {
            _Float16* dst = (p == 0 ? qb : kb);
#pragma unroll
            for (int j = 0; j < 8; j++)
                dst[((size_t)bh * T_ + t0 + j + 8 * half) * HD_ + e0 + n] =
                    (_Float16)c[j];
        } else {
            v8h pk;
#pragma unroll
            for (int j = 0; j < 8; j++) pk[j] = (_Float16)c[j];
            *(v8h*)(vtb + ((size_t)bh * HD_ + e0 + n) * T_ + t0 + 8 * half) = pk;
        }
    }
}

// ---------------------------------------------------------------------------
// Kernel 2: causal flash attention.  Block = 8 waves = 128 q-rows of one
// (b,h).  64-key K/V^T tiles (16KB) staged in LDS by async copies, shared by
// all 8 waves, in a TRIPLE-BUFFERED single-barrier pipeline: iteration i
// issues tile i+1 into buf[(i+1)%3], waits asynccnt<=4 (in-order completion
// => tile i landed), one barrier, then 16 WMMAs on tile i while tile i+1
// streams in.  All waves run the block-uniform tile count 2*(tb+1); compute
// is predicated on each wave's causal limit.
// ---------------------------------------------------------------------------
__global__ void k_attn(const _Float16* __restrict__ qb,
                       const _Float16* __restrict__ kb,
                       const _Float16* __restrict__ vtb,
                       _Float16* __restrict__ attb)
{
    // per buffer: K tile [64 key][64 d] (8KB) + V^T tile [64 d][64 s] (8KB)
    __shared__ __align__(16) _Float16 kvlds[3 * 8192];    // 48 KB
    __shared__ __align__(16) _Float16 plds[8 * 16 * 64];  // P staging 16 KB
    const int tid  = threadIdx.x;
    const int lane = tid & 31;
    const int wave = tid >> 5;
    const int bh = blockIdx.x >> 4;            // 64 (b,h) pairs
    const int tb = blockIdx.x & 15;            // 16 blocks of 128 rows
    const int b  = bh >> 4, h = bh & 15;
    const int t0 = tb * 128 + wave * 16;
    const int m    = lane & 15;
    const int n    = lane & 15;
    const int half = lane >> 4;

    // Q A-fragments (persistent)
    const _Float16* qrow = qb + ((size_t)bh * T_ + t0 + m) * HD_;
    v16h aq[2];
#pragma unroll
    for (int kd = 0; kd < 2; kd++) {
        v8h lo = *(const v8h*)(qrow + kd * 32 + half * 16);
        v8h hi = *(const v8h*)(qrow + kd * 32 + half * 16 + 8);
        aq[kd] = combine16(lo, hi);
    }

    const _Float16* kbase = kb + (size_t)bh * T_ * HD_;
    const _Float16* vbase = vtb + (size_t)bh * HD_ * T_;
    _Float16* lp = plds + wave * (16 * 64);
    const uint32_t kv0 = (uint32_t)(uintptr_t)&kvlds[0];
    // cooperative copy indices: 512 x 16B per region, 2 issues/thread/region
    const int vrow0 = tid >> 3, vseg0 = tid & 7;              // idx = tid
    const int vrow1 = (tid + 256) >> 3, vseg1 = tid & 7;      // idx = tid+256

    v8f o[4] = {{}, {}, {}, {}};
    float run_max[8], run_sum[8];
#pragma unroll
    for (int j = 0; j < 8; j++) { run_max[j] = -1e30f; run_sum[j] = 0.f; }

    const int nbk = 2 * (tb + 1);              // block-uniform tile count

    // issue one 16KB K/V tile (4 async copies per thread)
    auto issue_tile = [&](int kblk, int buf) {
        const int s0 = kblk * 64;
        const uint32_t kB = kv0 + (uint32_t)buf * (8192 * 2);
        const uint32_t vB = kB + 8192;
        // K tile: 8KB contiguous in kb
        const char* kg = (const char*)(kbase + (size_t)s0 * HD_);
        async_cp16(kB + tid * 16,         kg + tid * 16);
        async_cp16(kB + (tid + 256) * 16, kg + (tid + 256) * 16);
        // V^T tile: 64 rows of 128B, row-strided in vtb
        async_cp16(vB + (vrow0 * 64 + vseg0 * 8) * 2,
                   vbase + (size_t)vrow0 * T_ + s0 + vseg0 * 8);
        async_cp16(vB + (vrow1 * 64 + vseg1 * 8) * 2,
                   vbase + (size_t)vrow1 * T_ + s0 + vseg1 * 8);
    };

    issue_tile(0, 0);
    for (int kblk = 0; kblk < nbk; kblk++) {
        const int s0 = kblk * 64;
        if (kblk + 1 < nbk) { issue_tile(kblk + 1, (kblk + 1) % 3); async_wait4(); }
        else                { async_wait0(); }
        __syncthreads();                       // tile kblk visible to all

        if (s0 <= t0 + 15) {                   // this wave's causal window
            const _Float16* kt = kvlds + (kblk % 3) * 8192;
            const _Float16* vt = kt + 4096;

            // S = Q K^T : four 16x16 tiles, K=64 in two chunks
            v8f s[4];
#pragma unroll
            for (int ntile = 0; ntile < 4; ntile++) {
                v8f c = {};
#pragma unroll
                for (int kd = 0; kd < 2; kd++) {
                    v16h bf = *(const v16h*)(kt + (ntile * 16 + n) * HD_ +
                                             kd * 32 + half * 16);
                    c = wmma32(aq[kd], bf, c);
                }
                s[ntile] = c;
            }

            if (s0 + 63 > t0) {                // causal mask (partial tiles)
#pragma unroll
                for (int ntile = 0; ntile < 4; ntile++) {
                    const int sc = s0 + ntile * 16 + n;
#pragma unroll
                    for (int j = 0; j < 8; j++) {
                        const int t = t0 + j + 8 * half;
                        if (sc > t) s[ntile][j] = -1e30f;
                    }
                }
            }

            // online softmax: row stats reduced across the 16-lane group
            float alpha[8];
#pragma unroll
            for (int j = 0; j < 8; j++) {
                float mv = fmaxf(fmaxf(s[0][j], s[1][j]), fmaxf(s[2][j], s[3][j]));
                mv = fmaxf(mv, __shfl_xor(mv, 1));
                mv = fmaxf(mv, __shfl_xor(mv, 2));
                mv = fmaxf(mv, __shfl_xor(mv, 4));
                mv = fmaxf(mv, __shfl_xor(mv, 8));
                const float nm = fmaxf(run_max[j], mv);
                alpha[j] = __expf(run_max[j] - nm);
                run_max[j] = nm;
                const int row = j + 8 * half;
                float rs = 0.f;
#pragma unroll
                for (int ntile = 0; ntile < 4; ntile++) {
                    const float p = __expf(s[ntile][j] - nm);
                    lp[row * 64 + ntile * 16 + n] = (_Float16)p;
                    rs += p;
                }
                rs += __shfl_xor(rs, 1);
                rs += __shfl_xor(rs, 2);
                rs += __shfl_xor(rs, 4);
                rs += __shfl_xor(rs, 8);
                run_sum[j] = run_sum[j] * alpha[j] + rs;
            }
#pragma unroll
            for (int dt = 0; dt < 4; dt++)
#pragma unroll
                for (int j = 0; j < 8; j++) o[dt][j] *= alpha[j];

            // same-wave LDS RAW: DS pipe is in-order; fence compiler+counter
            asm volatile("s_wait_dscnt 0" ::: "memory");

            // P A-fragments (K=64 -> two 32-wide chunks)
            v16h pa[2];
#pragma unroll
            for (int c = 0; c < 2; c++) {
                v8h lo = *(const v8h*)(lp + m * 64 + c * 32 + half * 16);
                v8h hi = *(const v8h*)(lp + m * 64 + c * 32 + half * 16 + 8);
                pa[c] = combine16(lo, hi);
            }

            // O += P V  (V^T rows contiguous along s, from LDS)
#pragma unroll
            for (int dt = 0; dt < 4; dt++)
#pragma unroll
                for (int c = 0; c < 2; c++) {
                    v16h vb = *(const v16h*)(vt + (dt * 16 + n) * 64 +
                                             c * 32 + half * 16);
                    o[dt] = wmma32(pa[c], vb, o[dt]);
                }
        }
        // no trailing barrier: triple buffering keeps write target disjoint
        // from buffers any wave may still be reading this stage
    }

    // epilogue: normalize, write concat-head f16 output [B,T,C]
    float inv[8];
#pragma unroll
    for (int j = 0; j < 8; j++) inv[j] = 1.0f / run_sum[j];
    _Float16* arow0 = attb + ((size_t)b * T_ + t0) * C_ + h * HD_;
#pragma unroll
    for (int dt = 0; dt < 4; dt++)
#pragma unroll
        for (int j = 0; j < 8; j++)
            arow0[(size_t)(j + 8 * half) * C_ + dt * 16 + n] =
                (_Float16)(o[dt][j] * inv[j]);
}

// ---------------------------------------------------------------------------
// Kernel 3a: Wo fp32 -> f16 (rows already contiguous along K).
// ---------------------------------------------------------------------------
__global__ void k_cvt(const float* __restrict__ src, _Float16* __restrict__ dst,
                      int nElem)
{
    const int i = blockIdx.x * blockDim.x + threadIdx.x;
    if (i < nElem) dst[i] = (_Float16)src[i];
}

// ---------------------------------------------------------------------------
// Kernel 3: output projection Y = Att @ Wo^T + bo, register-blocked 32x64 per
// wave (2 m-tiles x 4 n-tiles): 6 fragment loads feed 8 WMMAs per K-step.
// ---------------------------------------------------------------------------
__global__ void k_proj(const _Float16* __restrict__ attb,
                       const _Float16* __restrict__ wob,
                       const float* __restrict__ bo,
                       float* __restrict__ out)
{
    const int lane = threadIdx.x & 31;
    const int wave = threadIdx.x >> 5;
    const int task = blockIdx.x * 8 + wave;    // 256 * 16 tasks
    const int nbk = task & 15;
    const int mbk = task >> 4;
    const int row0 = mbk * 32;
    const int n0   = nbk * 64;
    const int m    = lane & 15;
    const int n    = lane & 15;
    const int half = lane >> 4;

    const _Float16* arow0 = attb + (size_t)(row0 + m) * C_;
    const _Float16* arow1 = attb + (size_t)(row0 + 16 + m) * C_;
    v8f acc[2][4] = {{{}, {}, {}, {}}, {{}, {}, {}, {}}};
    for (int kc = 0; kc < 32; kc++) {          // K = 1024 in chunks of 32
        const int c0 = kc * 32;
        v16h a[2];
        a[0] = combine16(*(const v8h*)(arow0 + c0 + half * 16),
                         *(const v8h*)(arow0 + c0 + half * 16 + 8));
        a[1] = combine16(*(const v8h*)(arow1 + c0 + half * 16),
                         *(const v8h*)(arow1 + c0 + half * 16 + 8));
#pragma unroll
        for (int ni = 0; ni < 4; ni++) {
            v16h bfr = *(const v16h*)(wob + (size_t)(n0 + ni * 16 + n) * C_ +
                                      c0 + half * 16);
#pragma unroll
            for (int mi = 0; mi < 2; mi++)
                acc[mi][ni] = wmma32(a[mi], bfr, acc[mi][ni]);
        }
    }
#pragma unroll
    for (int ni = 0; ni < 4; ni++) {
        const float bias = bo[n0 + ni * 16 + n];
#pragma unroll
        for (int mi = 0; mi < 2; mi++)
#pragma unroll
            for (int j = 0; j < 8; j++)
                out[(size_t)(row0 + mi * 16 + j + 8 * half) * C_ +
                    n0 + ni * 16 + n] = acc[mi][ni][j] + bias;
    }
}

// ---------------------------------------------------------------------------
extern "C" void kernel_launch(void* const* d_in, const int* in_sizes, int n_in,
                              void* d_out, int out_size, void* d_ws, size_t ws_size,
                              hipStream_t stream)
{
    (void)in_sizes; (void)n_in; (void)out_size; (void)ws_size;
    const float* x  = (const float*)d_in[0];
    const float* Wq = (const float*)d_in[1];
    const float* Wk = (const float*)d_in[2];
    const float* Wv = (const float*)d_in[3];
    const float* Wo = (const float*)d_in[4];
    const float* bo = (const float*)d_in[5];
    float* out = (float*)d_out;

    // workspace layout (f16): q 16MB | k 16MB | v^T 16MB | att 16MB | Wo 2MB
    char* ws = (char*)d_ws;
    const size_t QK = (size_t)B_ * H_ * T_ * HD_;  // 8,388,608 elements
    _Float16* qb   = (_Float16*)(ws);
    _Float16* kb   = (_Float16*)(ws + QK * 2);
    _Float16* vtb  = (_Float16*)(ws + QK * 4);
    _Float16* attb = (_Float16*)(ws + QK * 6);
    _Float16* wob  = (_Float16*)(ws + QK * 8);

    k_cvt <<<(C_ * C_ + 255) / 256, 256, 0, stream>>>(Wo, wob, C_ * C_);
    k_qkv <<<(3 * 64 * 128) / 8,    256, 0, stream>>>(x, Wq, Wk, Wv, qb, kb, vtb);
    k_attn<<<64 * 16,               256, 0, stream>>>(qb, kb, vtb, attb);
    k_proj<<<(256 * 16) / 8,        256, 0, stream>>>(attb, wob, bo, out);
}
